// SelfAttention_8589934592338
// MI455X (gfx1250) — matrix operand order repaired and checked by reading, not compiled, not run on hardware
//
#include <hip/hip_runtime.h>

// Problem constants (match reference)
#define D_EMBED 1024
#define SEQ     2048
#define NHEADS  16
#define DHEAD   64
#define BATCH   2
#define MTOT    (BATCH * SEQ)      // 4096 tokens
#define N_QKV   (3 * D_EMBED)      // 3072

typedef __attribute__((ext_vector_type(16))) _Float16 v16h;
typedef __attribute__((ext_vector_type(8)))  _Float16 v8h;
typedef __attribute__((ext_vector_type(8)))  float    v8f;
typedef __attribute__((ext_vector_type(4)))  int      v4i;

#if __has_builtin(__builtin_amdgcn_wave_barrier)
#define WAVE_BARRIER() do { __builtin_amdgcn_wave_barrier(); asm volatile("" ::: "memory"); } while (0)
#else
#define WAVE_BARRIER() asm volatile("" ::: "memory")
#endif

#if __has_builtin(__builtin_amdgcn_s_wait_asynccnt)
#define WAIT_ASYNC(n) __builtin_amdgcn_s_wait_asynccnt(n)
#else
#define WAIT_ASYNC(n) asm volatile("s_wait_asynccnt " #n)
#endif

// Async copy of 16 halfs (32B) global -> LDS via two b128 async-to-LDS ops
// (tracked with ASYNCcnt).  Fallback: synchronous copy.
static __device__ inline void async_cp16h(const _Float16* g, _Float16* l) {
#if __has_builtin(__builtin_amdgcn_global_load_async_to_lds_b128)
  typedef __attribute__((address_space(1))) v4i* gp_t;   // global int4*
  typedef __attribute__((address_space(3))) v4i* lp_t;   // LDS int4*
  __builtin_amdgcn_global_load_async_to_lds_b128((gp_t)g, (lp_t)l, 0, 0);
  __builtin_amdgcn_global_load_async_to_lds_b128((gp_t)(g + 8), (lp_t)(l + 8), 0, 0);
#else
  *(v8h*)l = *(const v8h*)g;
  *(v8h*)(l + 8) = *(const v8h*)(g + 8);
#endif
}

// Load a 16-half fragment register set from two 16B chunks.
static __device__ inline v16h ld_16h(const _Float16* p0, const _Float16* p1) {
  v8h a = *(const v8h*)p0;
  v8h b = *(const v8h*)p1;
  v16h r;
#pragma unroll
  for (int i = 0; i < 8; ++i) { r[i] = a[i]; r[i + 8] = b[i]; }
  return r;
}

static __device__ inline v8f wmma_f16(v16h a, v16h b, v8f c) {
  // D = A(16x32 f16) x B(32x16 f16) + C(16x16 f32)
  return __builtin_amdgcn_wmma_f32_16x16x32_f16(false, a, false, b, (short)0, c,
                                                false, false);
}

// Reduce across the 16 lanes of each half-wave row group (wave32).
static __device__ inline float rowmax16(float v) {
#pragma unroll
  for (int m = 1; m <= 8; m <<= 1) v = fmaxf(v, __shfl_xor(v, m, 32));
  return v;
}
static __device__ inline float rowsum16(float v) {
#pragma unroll
  for (int m = 1; m <= 8; m <<= 1) v += __shfl_xor(v, m, 32);
  return v;
}

// ---------------------------------------------------------------- converts
__global__ void cvt_f32_to_f16(const float* __restrict__ src,
                               _Float16* __restrict__ dst, int n) {
  int i = blockIdx.x * blockDim.x + threadIdx.x;
  if (i < n) dst[i] = (_Float16)src[i];
}

// --------------------------------------------- block GEMM core (128x128)
// A: [M,K] f16 row-major.  Bw: [N,K] f16 row-major (torch Linear weights).
// 256 threads = 8 waves; wave w computes rows mb+(w>>1)*32, cols nb+(w&1)*64.
// B k-slices (128 cols x 32 k = 8KB) double-buffered in LDS via async-to-LDS.
static __device__ inline void gemm_block_core(const _Float16* __restrict__ A,
                                              const _Float16* __restrict__ Bw,
                                              _Float16 (*Bt)[128 * 32],
                                              int mb, int nb, int K,
                                              v8f acc[2][4]) {
  const int tid = threadIdx.x;
  const int w = tid >> 5, lane = tid & 31;
  const int grp = lane >> 4, cn = lane & 15;
  const int mw = mb + (w >> 1) * 32;
  const int nw = (w & 1) * 64;                 // col offset within block tile
  const int scol = tid >> 1, spart = (tid & 1) * 16;  // staging assignment
  const int akh = grp * 8;

  const int NCH = K / 32;
  // stage chunk 0 into buffer 0
  {
    const _Float16* g = Bw + (size_t)(nb + scol) * K + spart;
    async_cp16h(g, &Bt[0][scol * 32 + spart]);
  }
  // preload A fragments for chunk 0
  v16h a0, a1;
  {
    const _Float16* p0 = A + (size_t)(mw + cn) * K + akh;
    const _Float16* p1 = A + (size_t)(mw + 16 + cn) * K + akh;
    a0 = ld_16h(p0, p0 + 16);
    a1 = ld_16h(p1, p1 + 16);
  }

  for (int ch = 0; ch < NCH; ++ch) {
    const int buf = ch & 1;
    const bool more = (ch + 1) < NCH;
    if (more) {  // stage next chunk into other buffer
      const _Float16* g = Bw + (size_t)(nb + scol) * K + (ch + 1) * 32 + spart;
      async_cp16h(g, &Bt[buf ^ 1][scol * 32 + spart]);
    }
    if (more) { WAIT_ASYNC(2); } else { WAIT_ASYNC(0); }
    __syncthreads();

    v16h na0 = a0, na1 = a1;
    if (more) {  // software-pipeline next A fragments
      const _Float16* p0 = A + (size_t)(mw + cn) * K + (ch + 1) * 32 + akh;
      const _Float16* p1 = A + (size_t)(mw + 16 + cn) * K + (ch + 1) * 32 + akh;
      __builtin_prefetch(p0 + 64, 0, 3);
      na0 = ld_16h(p0, p0 + 16);
      na1 = ld_16h(p1, p1 + 16);
    }
#pragma unroll
    for (int j = 0; j < 4; ++j) {
      const _Float16* bp = &Bt[buf][(nw + j * 16 + cn) * 32 + grp * 8];
      v16h b = ld_16h(bp, bp + 16);
      acc[0][j] = wmma_f16(a0, b, acc[0][j]);
      acc[1][j] = wmma_f16(a1, b, acc[1][j]);
    }
    __syncthreads();   // protect buffer reuse before next stage
    a0 = na0; a1 = na1;
  }
}

// ------------------------------------------------------------- QKV GEMM
// qkv[m,n] = sum_k x[m,k]*w_in[n,k] + b_in[n]; route into qh / kh / vt (f16).
// vt layout: [b, h, dh, s] so PV B-fragments are contiguous along keys.
__global__ void __launch_bounds__(256)
qkv_gemm_kernel(const _Float16* __restrict__ xh,
                const _Float16* __restrict__ w_inh,
                const float* __restrict__ b_in,
                _Float16* __restrict__ qh,
                _Float16* __restrict__ kh,
                _Float16* __restrict__ vt) {
  __shared__ __align__(16) _Float16 Bt[2][128 * 32];
  const int ntN = N_QKV / 128;                 // 24
  const int mb = (blockIdx.x / ntN) * 128;
  const int nb = (blockIdx.x % ntN) * 128;
  v8f acc[2][4] = {};
  gemm_block_core(xh, w_inh, Bt, mb, nb, D_EMBED, acc);

  const int tid = threadIdx.x, w = tid >> 5, lane = tid & 31;
  const int grp = lane >> 4, cn = lane & 15;
  const int mw = mb + (w >> 1) * 32;
  const int n0 = nb + (w & 1) * 64;
#pragma unroll
  for (int fm = 0; fm < 2; ++fm)
#pragma unroll
    for (int j = 0; j < 4; ++j)
#pragma unroll
      for (int r = 0; r < 8; ++r) {
        const int m = mw + fm * 16 + r + grp * 8;
        const int n = n0 + j * 16 + cn;
        const _Float16 hv = (_Float16)(acc[fm][j][r] + b_in[n]);
        if (n < D_EMBED) {
          qh[(size_t)m * D_EMBED + n] = hv;
        } else if (n < 2 * D_EMBED) {
          kh[(size_t)m * D_EMBED + (n - D_EMBED)] = hv;
        } else {
          const int nv = n - 2 * D_EMBED;      // h*64 + d
          const int bb = m >> 11, ss = m & (SEQ - 1);
          vt[((size_t)bb * D_EMBED + nv) * SEQ + ss] = hv;
        }
      }
}

// --------------------------------------------------- flash attention kernel
// One wave handles (b, h, 16-query tile); online softmax over 32-key chunks.
__global__ void __launch_bounds__(128)
attn_kernel(const _Float16* __restrict__ qh,
            const _Float16* __restrict__ kh,
            const _Float16* __restrict__ vt,
            _Float16* __restrict__ oh,
            const int* __restrict__ causal_ptr) {
  __shared__ __align__(16) _Float16 ptile[4][16 * 32];  // per-wave P re-layout
  const int causal = *causal_ptr;
  const int w = threadIdx.x >> 5;
  const int wid = blockIdx.x * 4 + w;
  const int QT = SEQ / 16;               // 128 query tiles per (b,h)
  const int qt = wid % QT;
  const int h  = (wid / QT) & (NHEADS - 1);
  const int b  = wid / (QT * NHEADS);
  const int q0 = qt * 16;
  const int lane = threadIdx.x & 31;
  const int grp = lane >> 4, cn = lane & 15;

  // Q A-fragments (held in registers for whole tile), pre-scaled by 1/sqrt(dh)
  v16h qa[2];
  {
    const _Float16* qrow = qh + ((size_t)(b * SEQ + q0 + cn)) * D_EMBED + h * DHEAD;
#pragma unroll
    for (int c = 0; c < 2; ++c) {
      const _Float16* p = qrow + c * 32 + grp * 8;
      qa[c] = ld_16h(p, p + 16);
#pragma unroll
      for (int i = 0; i < 16; ++i) qa[c][i] = qa[c][i] * (_Float16)0.125f;
    }
  }

  float mrow[8], lrow[8];
  v8f o[4] = {};
#pragma unroll
  for (int r = 0; r < 8; ++r) { mrow[r] = -__builtin_inff(); lrow[r] = 0.0f; }

  const _Float16* Kbase = kh + ((size_t)b * SEQ) * D_EMBED + h * DHEAD;
  const _Float16* Vbase = vt + ((size_t)b * D_EMBED + h * DHEAD) * SEQ;
  const int kmax = causal ? (q0 + 15) : (SEQ - 1);
  const int nch = kmax / 32 + 1;

  for (int ch = 0; ch < nch; ++ch) {
    const int kb0 = ch * 32;
    // ---- scores: 2 fragments of 16 keys, each = 2 WMMAs over dh
    v8f s[2];
#pragma unroll
    for (int f = 0; f < 2; ++f) {
      v8f z = {};
      const int key = kb0 + f * 16 + cn;
#pragma unroll
      for (int c = 0; c < 2; ++c) {
        const _Float16* p = Kbase + (size_t)key * D_EMBED + c * 32 + grp * 8;
        v16h kf = ld_16h(p, p + 16);
        z = wmma_f16(qa[c], kf, z);
      }
      if (causal) {
#pragma unroll
        for (int r = 0; r < 8; ++r)
          if (kb0 + f * 16 + cn > q0 + r + grp * 8) z[r] = -__builtin_inff();
      }
      s[f] = z;
    }
    // ---- online softmax update per row
    float cfac[8];
#pragma unroll
    for (int r = 0; r < 8; ++r) {
      float mx = rowmax16(fmaxf(s[0][r], s[1][r]));
      const float mnew = fmaxf(mrow[r], mx);
      cfac[r] = __expf(mrow[r] - mnew);
      mrow[r] = mnew;
      const float p0 = __expf(s[0][r] - mnew);
      const float p1 = __expf(s[1][r] - mnew);
      s[0][r] = p0; s[1][r] = p1;
      lrow[r] = lrow[r] * cfac[r] + rowsum16(p0 + p1);
    }
#pragma unroll
    for (int j = 0; j < 4; ++j)
#pragma unroll
      for (int r = 0; r < 8; ++r) o[j][r] *= cfac[r];

    // ---- re-layout P (C layout -> A layout) via wave-private LDS tile
#pragma unroll
    for (int f = 0; f < 2; ++f)
#pragma unroll
      for (int r = 0; r < 8; ++r)
        ptile[w][(r + grp * 8) * 32 + f * 16 + cn] = (_Float16)s[f][r];
    WAVE_BARRIER();
    const _Float16* pp = &ptile[w][cn * 32 + grp * 8];
    v16h pa = ld_16h(pp, pp + 16);
    WAVE_BARRIER();

    // ---- O += P * V  (4 dh-column fragments)
#pragma unroll
    for (int j = 0; j < 4; ++j) {
      const _Float16* vp = Vbase + (size_t)(j * 16 + cn) * SEQ + kb0 + grp * 8;
      v16h vb = ld_16h(vp, vp + 16);
      o[j] = wmma_f16(pa, vb, o[j]);
    }
  }

  // ---- normalize and write O (f16, token-major [B*S, D])
  float inv[8];
#pragma unroll
  for (int r = 0; r < 8; ++r) inv[r] = 1.0f / lrow[r];
#pragma unroll
  for (int j = 0; j < 4; ++j)
#pragma unroll
    for (int r = 0; r < 8; ++r) {
      const int m = b * SEQ + q0 + r + grp * 8;
      oh[(size_t)m * D_EMBED + h * DHEAD + j * 16 + cn] = (_Float16)(o[j][r] * inv[r]);
    }
}

// ------------------------------------------------------------ output GEMM
__global__ void __launch_bounds__(256)
out_gemm_kernel(const _Float16* __restrict__ ohh,
                const _Float16* __restrict__ w_outh,
                const float* __restrict__ b_out,
                float* __restrict__ out) {
  __shared__ __align__(16) _Float16 Bt[2][128 * 32];
  const int ntN = D_EMBED / 128;               // 8
  const int mb = (blockIdx.x / ntN) * 128;
  const int nb = (blockIdx.x % ntN) * 128;
  v8f acc[2][4] = {};
  gemm_block_core(ohh, w_outh, Bt, mb, nb, D_EMBED, acc);

  const int tid = threadIdx.x, w = tid >> 5, lane = tid & 31;
  const int grp = lane >> 4, cn = lane & 15;
  const int mw = mb + (w >> 1) * 32;
  const int n0 = nb + (w & 1) * 64;
#pragma unroll
  for (int fm = 0; fm < 2; ++fm)
#pragma unroll
    for (int j = 0; j < 4; ++j)
#pragma unroll
      for (int r = 0; r < 8; ++r) {
        const int m = mw + fm * 16 + r + grp * 8;
        const int n = n0 + j * 16 + cn;
        out[(size_t)m * D_EMBED + n] = acc[fm][j][r] + b_out[n];
      }
}

// ----------------------------------------------------------------- launch
extern "C" void kernel_launch(void* const* d_in, const int* in_sizes, int n_in,
                              void* d_out, int out_size, void* d_ws, size_t ws_size,
                              hipStream_t stream) {
  (void)in_sizes; (void)n_in; (void)out_size; (void)ws_size;
  const float* x      = (const float*)d_in[0];
  const float* w_in   = (const float*)d_in[1];
  const float* b_in   = (const float*)d_in[2];
  const float* w_out  = (const float*)d_in[3];
  const float* b_out  = (const float*)d_in[4];
  const int*   causal = (const int*)d_in[5];
  float* out = (float*)d_out;

  // Workspace carve (f16): ~48 MB total
  _Float16* ws     = (_Float16*)d_ws;
  const size_t M4  = (size_t)MTOT * D_EMBED;        // 4096*1024
  _Float16* xh     = ws;
  _Float16* w_inh  = xh + M4;                       // 3072*1024
  _Float16* w_outh = w_inh + (size_t)N_QKV * D_EMBED;
  _Float16* qh     = w_outh + (size_t)D_EMBED * D_EMBED;
  _Float16* kh     = qh + M4;
  _Float16* vt     = kh + M4;
  _Float16* oh     = vt + M4;

  const int nx = MTOT * D_EMBED;
  const int nwi = N_QKV * D_EMBED;
  const int nwo = D_EMBED * D_EMBED;
  cvt_f32_to_f16<<<(nx + 255) / 256, 256, 0, stream>>>(x, xh, nx);
  cvt_f32_to_f16<<<(nwi + 255) / 256, 256, 0, stream>>>(w_in, w_inh, nwi);
  cvt_f32_to_f16<<<(nwo + 255) / 256, 256, 0, stream>>>(w_out, w_outh, nwo);

  // QKV: (4096/128)*(3072/128) = 32*24 = 768 block tiles
  qkv_gemm_kernel<<<768, 256, 0, stream>>>(xh, w_inh, b_in, qh, kh, vt);
  // Attention: 2*16*128 = 4096 waves, 4 waves/block
  attn_kernel<<<1024, 128, 0, stream>>>(qh, kh, vt, oh, causal);
  // Output projection: (4096/128)*(1024/128) = 32*8 = 256 block tiles
  out_gemm_kernel<<<256, 256, 0, stream>>>(oh, w_outh, b_out, out);
}
